// NeuralHP_9053791060150
// MI455X (gfx1250) — matrix operand order repaired and checked by reading, not compiled, not run on hardware
//
#include <hip/hip_runtime.h>
#include <math.h>

typedef __attribute__((ext_vector_type(16))) _Float16 v16h;
typedef __attribute__((ext_vector_type(8)))  _Float16 v8h;
typedef __attribute__((ext_vector_type(4)))  _Float16 v4h;
typedef __attribute__((ext_vector_type(8)))  float    v8f;
typedef _Float16 h16;

static constexpr int Nn = 8192;   // nodes
static constexpr int Mm = 8192;   // hyperlinks
static constexpr int Dd = 256;    // feature dim
static constexpr int Hh = 128;    // hidden dim

static constexpr int KSLAB  = 64;   // K-chunk staged to LDS (2 WMMA k-steps)
static constexpr int BPITCH = 72;   // padded LDS row pitch (halves) -> conflict-free b128 reads

// CDNA5 async global->LDS DMA path (ASYNCcnt-tracked), if the toolchain has it.
#if defined(__gfx1250__) && \
    __has_builtin(__builtin_amdgcn_global_load_async_to_lds_b128) && \
    __has_builtin(__builtin_amdgcn_s_wait_asynccnt)
#define USE_ASYNC_LDS 1
#else
#define USE_ASYNC_LDS 0
#endif

// ---------------------------------------------------------------------------
// WMMA core: one K-slab. A tile (16 rows x KSLAB) read from global (row-major,
// lda in halves); B slab ([128 h-rows][KSLAB]) read from LDS (column-major of
// the logical B, i.e. lane = column). 8 column tiles cover full H=128.
// ---------------------------------------------------------------------------
__device__ __forceinline__ void wmma_slab(const h16* __restrict__ Atile, size_t lda,
                                          const h16* __restrict__ ldsB,
                                          int lane, v8f acc[8]) {
  const int col = lane & 15;
  const int kb  = (lane & 16) ? 8  : 0;   // A-fragment K base (low half)
  const int khi = (lane & 16) ? 8 : 0;    // B-fragment K base
#pragma unroll
  for (int kk = 0; kk < KSLAB / 32; ++kk) {
    const h16* pa = Atile + (size_t)col * lda + kk * 32 + kb;
    v8h alo = *(const v8h*)pa;
    v8h ahi = *(const v8h*)(pa + 16);
    v16h a;
#pragma unroll
    for (int i = 0; i < 8; ++i) { a[i] = alo[i]; a[i + 8] = ahi[i]; }
#pragma unroll
    for (int t = 0; t < 8; ++t) {
      const h16* pb = ldsB + (size_t)(t * 16 + col) * BPITCH + kk * 32 + khi;
      v8h blo = *(const v8h*)pb;
      v8h bhi = *(const v8h*)(pb + 16);
      v16h bf;
#pragma unroll
      for (int i = 0; i < 8; ++i) { bf[i] = blo[i]; bf[i + 8] = bhi[i]; }
      acc[t] = __builtin_amdgcn_wmma_f32_16x16x32_f16(
          false, a, false, bf, (short)0, acc[t], false, false);
    }
  }
}

// ---------------------------------------------------------------------------
// B-slab staging: 128 rows x KSLAB halves, 256 threads, 32 B per thread.
// Async path: global_load_async_to_lds_b128 (no VGPR staging, ASYNCcnt).
// Fallback: register-staged global->VGPR->ds_store copy.
// ---------------------------------------------------------------------------
#if USE_ASYNC_LDS
typedef int v4i_ __attribute__((__vector_size__(16)));
typedef __attribute__((address_space(1))) v4i_ g_v4i;
typedef __attribute__((address_space(3))) v4i_ l_v4i;

__device__ __forceinline__ void stage_async(const h16* __restrict__ Bb, size_t ldb,
                                            int k0, int tid, h16* lds) {
  const h16* src = Bb + (size_t)(tid >> 1) * ldb + k0 + (tid & 1) * 32;
  h16* dst = lds + (size_t)(tid >> 1) * BPITCH + (tid & 1) * 32;
  g_v4i* g = (g_v4i*)src;
  l_v4i* l = (l_v4i*)dst;
  __builtin_amdgcn_global_load_async_to_lds_b128(g, l, 0, 0);
  __builtin_amdgcn_global_load_async_to_lds_b128(g, l, 16, 0);  // offset adds to both sides
}
#define STAGE_WAIT() __builtin_amdgcn_s_wait_asynccnt(0)
#else
__device__ __forceinline__ void stage_load(const h16* __restrict__ Bb, size_t ldb,
                                           int k0, int tid, uint4 st[2]) {
  const uint4* src = (const uint4*)(Bb + (size_t)(tid >> 1) * ldb + k0 + (tid & 1) * 32);
  st[0] = src[0];
  st[1] = src[1];
}
__device__ __forceinline__ void stage_store(h16* lds, int tid, const uint4 st[2]) {
  uint4* dst = (uint4*)(lds + (size_t)(tid >> 1) * BPITCH + (tid & 1) * 32);
  dst[0] = st[0];
  dst[1] = st[1];
}
#endif

// ---------------------------------------------------------------------------
// Elementwise f32 -> f16 (vectorized x4)
// ---------------------------------------------------------------------------
__global__ void cvt_f32_f16(const float* __restrict__ src, h16* __restrict__ dst, int n4) {
  int i = blockIdx.x * blockDim.x + threadIdx.x;
  if (i < n4) {
    float4 v = ((const float4*)src)[i];
    v4h o;
    o[0] = (h16)v.x; o[1] = (h16)v.y; o[2] = (h16)v.z; o[3] = (h16)v.w;
    ((v4h*)dst)[i] = o;
  }
}

// loop_w,gcn_w [D,H] f32 -> transposed f16 [H,D]
__global__ void cvt_weightsT(const float* __restrict__ lw, const float* __restrict__ gw,
                             h16* __restrict__ lwT, h16* __restrict__ gwT) {
  int t = blockIdx.x * blockDim.x + threadIdx.x;
  const int DH = Dd * Hh;
  if (t < DH) {
    int h = t / Dd, d = t % Dd;
    lwT[t] = (h16)lw[(size_t)d * Hh + h];
  } else if (t < 2 * DH) {
    int u = t - DH;
    int h = u / Dd, d = u % Dd;
    gwT[u] = (h16)gw[(size_t)d * Hh + h];
  }
}

// I [N,M] f32 -> I16 [N,M] f16, IT16 [M,N] f16, cnt[m] += column sums
__global__ __launch_bounds__(256) void transpose_I(const float* __restrict__ I,
                                                   h16* __restrict__ I16,
                                                   h16* __restrict__ IT16,
                                                   float* __restrict__ cnt) {
  __shared__ float tile[32][33];
  const int tx = threadIdx.x, ty = threadIdx.y;
  const int m0 = blockIdx.x * 32, n0 = blockIdx.y * 32;
#pragma unroll
  for (int k = 0; k < 4; ++k) {
    int rr = ty + k * 8;
    float v = I[(size_t)(n0 + rr) * Mm + m0 + tx];
    tile[rr][tx] = v;
    I16[(size_t)(n0 + rr) * Mm + m0 + tx] = (h16)v;
  }
  __syncthreads();
#pragma unroll
  for (int k = 0; k < 4; ++k) {
    int rr = ty + k * 8;
    IT16[(size_t)(m0 + rr) * Nn + n0 + tx] = (h16)tile[tx][rr];
  }
  if (ty == 0) {
    float s = 0.f;
#pragma unroll
    for (int r = 0; r < 32; ++r) s += tile[r][tx];
    atomicAdd(&cnt[m0 + tx], s);
  }
}

// ---------------------------------------------------------------------------
// buildG: GT[h,m] = sum_d AX16[m,d]*gwT[h,d] + gcn_b[h]   (transposed f16 out)
// ---------------------------------------------------------------------------
__global__ __launch_bounds__(256, 2) void buildG(const h16* __restrict__ iAX16,
                                                 const h16* __restrict__ jAX16,
                                                 const h16* __restrict__ gwT,
                                                 const float* __restrict__ gcn_b,
                                                 h16* __restrict__ GTi,
                                                 h16* __restrict__ GTj) {
  __shared__ h16 lds[2][128 * BPITCH];
  const int tid = threadIdx.x, lane = tid & 31, w = tid >> 5;
  const int side = blockIdx.y;
  const h16* AX = side ? jAX16 : iAX16;
  h16* GT = side ? GTj : GTi;
  const int m0 = blockIdx.x * 128 + w * 16;

  v8f acc[8];
#pragma unroll
  for (int t = 0; t < 8; ++t)
#pragma unroll
    for (int i = 0; i < 8; ++i) acc[t][i] = 0.f;

  const int NSL = Dd / KSLAB;  // 4
  const h16* Abase = AX + (size_t)m0 * Dd;
  int buf = 0;

#if USE_ASYNC_LDS
  stage_async(gwT, Dd, 0, tid, lds[0]);
  STAGE_WAIT();
  __syncthreads();
  for (int s = 0; s < NSL; ++s) {
    const bool pf = (s + 1 < NSL);
    if (pf) stage_async(gwT, Dd, (s + 1) * KSLAB, tid, lds[buf ^ 1]);
    wmma_slab(Abase + s * KSLAB, Dd, lds[buf], lane, acc);
    if (pf) STAGE_WAIT();
    __syncthreads();
    buf ^= 1;
  }
#else
  {
    uint4 st[2];
    stage_load(gwT, Dd, 0, tid, st);
    stage_store(lds[0], tid, st);
  }
  __syncthreads();
  for (int s = 0; s < NSL; ++s) {
    uint4 st[2];
    bool pf = (s + 1 < NSL);
    if (pf) stage_load(gwT, Dd, (s + 1) * KSLAB, tid, st);
    wmma_slab(Abase + s * KSLAB, Dd, lds[buf], lane, acc);
    if (pf) stage_store(lds[buf ^ 1], tid, st);
    __syncthreads();
    buf ^= 1;
  }
#endif

  const int col = lane & 15, hi4 = (lane >> 4);
#pragma unroll
  for (int t = 0; t < 8; ++t) {
    int h = t * 16 + col;
    float gb = gcn_b[h];
    v8h o;
#pragma unroll
    for (int i = 0; i < 8; ++i) o[i] = (h16)(acc[t][i] + gb);
    *(v8h*)(GT + (size_t)h * Mm + m0 + 8 * hi4) = o;
  }
}

// ---------------------------------------------------------------------------
// gemm1: HT[h,n] = loop_b[h] + sum_m I16[n,m]*GT[h,m] + sum_d X16[n,d]*lwT[h,d]
// (concatenated-K GEMM, transposed f16 output)
// ---------------------------------------------------------------------------
__global__ __launch_bounds__(256, 2) void gemm1(const h16* __restrict__ I16,
                                                const h16* __restrict__ iX16,
                                                const h16* __restrict__ jX16,
                                                const h16* __restrict__ GTi,
                                                const h16* __restrict__ GTj,
                                                const h16* __restrict__ lwT,
                                                const float* __restrict__ loop_b,
                                                h16* __restrict__ HTi,
                                                h16* __restrict__ HTj) {
  __shared__ h16 lds[2][128 * BPITCH];
  const int tid = threadIdx.x, lane = tid & 31, w = tid >> 5;
  const int side = blockIdx.y;
  const h16* X16 = side ? jX16 : iX16;
  const h16* GT  = side ? GTj : GTi;
  h16* HT        = side ? HTj : HTi;
  const int n0 = blockIdx.x * 128 + w * 16;
  const int col = lane & 15, hi4 = (lane >> 4);

  v8f acc[8];
#pragma unroll
  for (int t = 0; t < 8; ++t) {
    float b = loop_b[t * 16 + col];
#pragma unroll
    for (int i = 0; i < 8; ++i) acc[t][i] = b;
  }

  const h16* Aseg1 = I16 + (size_t)n0 * Mm;
  const h16* Aseg2 = X16 + (size_t)n0 * Dd;
  const int NS1 = Mm / KSLAB;          // 128
  const int NS  = NS1 + Dd / KSLAB;    // +4

  auto getseg = [&](int s, const h16*& A, size_t& lda,
                    const h16*& B, size_t& ldb, int& k0) {
    if (s < NS1) { A = Aseg1; lda = Mm; B = GT;  ldb = Mm; k0 = s * KSLAB; }
    else         { A = Aseg2; lda = Dd; B = lwT; ldb = Dd; k0 = (s - NS1) * KSLAB; }
  };

  int buf = 0;
#if USE_ASYNC_LDS
  {
    const h16 *A, *B; size_t lda, ldb; int k0;
    getseg(0, A, lda, B, ldb, k0);
    stage_async(B, ldb, k0, tid, lds[0]);
    STAGE_WAIT();
  }
  __syncthreads();
  for (int s = 0; s < NS; ++s) {
    const bool pf = (s + 1 < NS);
    if (pf) {
      const h16 *An, *Bn; size_t ldan, ldbn; int k0n;
      getseg(s + 1, An, ldan, Bn, ldbn, k0n);
      stage_async(Bn, ldbn, k0n, tid, lds[buf ^ 1]);
      __builtin_prefetch(An + k0n + (size_t)col * ldan, 0, 3);  // -> global_prefetch_b8
    }
    const h16 *Ac, *Bc; size_t ldac, ldbc; int k0c;
    getseg(s, Ac, ldac, Bc, ldbc, k0c);
    wmma_slab(Ac + k0c, ldac, lds[buf], lane, acc);
    if (pf) STAGE_WAIT();
    __syncthreads();
    buf ^= 1;
  }
#else
  {
    const h16 *A, *B; size_t lda, ldb; int k0;
    getseg(0, A, lda, B, ldb, k0);
    uint4 st[2];
    stage_load(B, ldb, k0, tid, st);
    stage_store(lds[0], tid, st);
  }
  __syncthreads();
  for (int s = 0; s < NS; ++s) {
    uint4 st[2];
    const bool pf = (s + 1 < NS);
    if (pf) {
      const h16 *An, *Bn; size_t ldan, ldbn; int k0n;
      getseg(s + 1, An, ldan, Bn, ldbn, k0n);
      stage_load(Bn, ldbn, k0n, tid, st);
      __builtin_prefetch(An + k0n + (size_t)col * ldan, 0, 3);
    }
    const h16 *Ac, *Bc; size_t ldac, ldbc; int k0c;
    getseg(s, Ac, ldac, Bc, ldbc, k0c);
    wmma_slab(Ac + k0c, ldac, lds[buf], lane, acc);
    if (pf) stage_store(lds[buf ^ 1], tid, st);
    __syncthreads();
    buf ^= 1;
  }
#endif

#pragma unroll
  for (int t = 0; t < 8; ++t) {
    int h = t * 16 + col;
    v8h o;
#pragma unroll
    for (int i = 0; i < 8; ++i) o[i] = (h16)acc[t][i];
    *(v8h*)(HT + (size_t)h * Nn + n0 + 8 * hi4) = o;
  }
}

// ---------------------------------------------------------------------------
// gemm2: raw[m,h] = sum_n IT16[m,n]*HT[h,n]; fused epilogue:
//        out[m] = sigmoid( (raw[m,:] . int_w) / cnt[m] + int_b )
// ---------------------------------------------------------------------------
__global__ __launch_bounds__(256, 2) void gemm2(const h16* __restrict__ IT16,
                                                const h16* __restrict__ HTi,
                                                const h16* __restrict__ HTj,
                                                const float* __restrict__ int_w,
                                                const float* __restrict__ int_b,
                                                const float* __restrict__ cnt,
                                                float* __restrict__ out_base) {
  __shared__ h16 lds[2][128 * BPITCH];
  const int tid = threadIdx.x, lane = tid & 31, w = tid >> 5;
  const int side = blockIdx.y;
  const h16* HT = side ? HTj : HTi;
  float* out = out_base + (size_t)side * Mm;
  const int m0 = blockIdx.x * 128 + w * 16;
  const int col = lane & 15, hi4 = (lane >> 4);

  v8f acc[8];
#pragma unroll
  for (int t = 0; t < 8; ++t)
#pragma unroll
    for (int i = 0; i < 8; ++i) acc[t][i] = 0.f;

  const h16* Abase = IT16 + (size_t)m0 * Nn;
  const int NS = Nn / KSLAB;  // 128
  int buf = 0;

#if USE_ASYNC_LDS
  stage_async(HT, Nn, 0, tid, lds[0]);
  STAGE_WAIT();
  __syncthreads();
  for (int s = 0; s < NS; ++s) {
    const bool pf = (s + 1 < NS);
    if (pf) {
      stage_async(HT, Nn, (s + 1) * KSLAB, tid, lds[buf ^ 1]);
      __builtin_prefetch(Abase + (s + 1) * KSLAB + (size_t)col * Nn, 0, 3);
    }
    wmma_slab(Abase + s * KSLAB, Nn, lds[buf], lane, acc);
    if (pf) STAGE_WAIT();
    __syncthreads();
    buf ^= 1;
  }
#else
  {
    uint4 st[2];
    stage_load(HT, Nn, 0, tid, st);
    stage_store(lds[0], tid, st);
  }
  __syncthreads();
  for (int s = 0; s < NS; ++s) {
    uint4 st[2];
    const bool pf = (s + 1 < NS);
    if (pf) {
      stage_load(HT, Nn, (s + 1) * KSLAB, tid, st);
      __builtin_prefetch(Abase + (s + 1) * KSLAB + (size_t)col * Nn, 0, 3);
    }
    wmma_slab(Abase + s * KSLAB, Nn, lds[buf], lane, acc);
    if (pf) stage_store(lds[buf ^ 1], tid, st);
    __syncthreads();
    buf ^= 1;
  }
#endif

  // fused epilogue: dot with int_w, scale by 1/cnt, sigmoid
  float wv[8];
#pragma unroll
  for (int t = 0; t < 8; ++t) wv[t] = int_w[t * 16 + col];
  const float bint = int_b[0];
#pragma unroll
  for (int r = 0; r < 8; ++r) {
    float p = 0.f;
#pragma unroll
    for (int t = 0; t < 8; ++t) p += acc[t][r] * wv[t];
    // reduce across the 16 lanes of each half-wave (xor<16 keeps halves apart)
    for (int off = 1; off < 16; off <<= 1) p += __shfl_xor(p, off, 32);
    if (col == 0) {
      int m = m0 + r + 8 * hi4;
      float z = p * (1.0f / cnt[m]) + bint;
      out[m] = 1.0f / (1.0f + __expf(-z));
    }
  }
}

// ---------------------------------------------------------------------------
// launcher
// ---------------------------------------------------------------------------
static inline size_t align_up(size_t x, size_t a) { return (x + a - 1) & ~(a - 1); }

extern "C" void kernel_launch(void* const* d_in, const int* in_sizes, int n_in,
                              void* d_out, int out_size, void* d_ws, size_t ws_size,
                              hipStream_t stream) {
  const float* iX     = (const float*)d_in[0];
  const float* jX     = (const float*)d_in[1];
  const float* iAX    = (const float*)d_in[2];
  const float* jAX    = (const float*)d_in[3];
  const float* I      = (const float*)d_in[4];
  const float* loop_w = (const float*)d_in[5];
  const float* loop_b = (const float*)d_in[6];
  const float* gcn_w  = (const float*)d_in[7];
  const float* gcn_b  = (const float*)d_in[8];
  const float* int_w  = (const float*)d_in[9];
  const float* int_b  = (const float*)d_in[10];

  char* ws = (char*)d_ws;
  size_t off = 0;
  auto carve = [&](size_t bytes) {
    void* p = ws + off;
    off = align_up(off + bytes, 256);
    return p;
  };
  h16* I16    = (h16*)carve((size_t)Nn * Mm * 2);
  h16* IT16   = (h16*)carve((size_t)Mm * Nn * 2);
  h16* iX16   = (h16*)carve((size_t)Nn * Dd * 2);
  h16* jX16   = (h16*)carve((size_t)Nn * Dd * 2);
  h16* iAX16  = (h16*)carve((size_t)Mm * Dd * 2);
  h16* jAX16  = (h16*)carve((size_t)Mm * Dd * 2);
  h16* lwT    = (h16*)carve((size_t)Hh * Dd * 2);
  h16* gwT    = (h16*)carve((size_t)Hh * Dd * 2);
  h16* GTi    = (h16*)carve((size_t)Hh * Mm * 2);
  h16* GTj    = (h16*)carve((size_t)Hh * Mm * 2);
  h16* HTi    = (h16*)carve((size_t)Hh * Nn * 2);
  h16* HTj    = (h16*)carve((size_t)Hh * Nn * 2);
  float* cnt  = (float*)carve((size_t)Mm * 4);
  (void)ws_size; (void)in_sizes; (void)n_in; (void)out_size;

  (void)hipMemsetAsync(cnt, 0, (size_t)Mm * 4, stream);

  const int n4 = Nn * Dd / 4;
  dim3 cgrid((n4 + 255) / 256);
  cvt_f32_f16<<<cgrid, 256, 0, stream>>>(iX,  iX16,  n4);
  cvt_f32_f16<<<cgrid, 256, 0, stream>>>(jX,  jX16,  n4);
  cvt_f32_f16<<<cgrid, 256, 0, stream>>>(iAX, iAX16, n4);
  cvt_f32_f16<<<cgrid, 256, 0, stream>>>(jAX, jAX16, n4);

  cvt_weightsT<<<(2 * Dd * Hh + 255) / 256, 256, 0, stream>>>(loop_w, gcn_w, lwT, gwT);

  transpose_I<<<dim3(Mm / 32, Nn / 32), dim3(32, 8), 0, stream>>>(I, I16, IT16, cnt);

  buildG<<<dim3(Mm / 128, 2), 256, 0, stream>>>(iAX16, jAX16, gwT, gcn_b, GTi, GTj);

  gemm1<<<dim3(Nn / 128, 2), 256, 0, stream>>>(I16, iX16, jX16, GTi, GTj, lwT, loop_b,
                                               HTi, HTj);

  gemm2<<<dim3(Mm / 128, 2), 256, 0, stream>>>(IT16, HTi, HTj, int_w, int_b, cnt,
                                               (float*)d_out);
}
